// MultiHeadedAttention_55765855371402
// MI455X (gfx1250) — matrix operand order repaired and checked by reading, not compiled, not run on hardware
//
#include <hip/hip_runtime.h>
#include <hip/hip_bf16.h>

typedef __attribute__((ext_vector_type(16))) __bf16 v16bf;
typedef __attribute__((ext_vector_type(8)))  __bf16 v8bf;
typedef __attribute__((ext_vector_type(8)))  float  v8f;
typedef __attribute__((ext_vector_type(4)))  unsigned int v4u;
typedef __attribute__((ext_vector_type(8)))  int v8i_;
typedef __attribute__((ext_vector_type(4)))  int v4i_;

constexpr int kB = 2, kS = 2048, kD = 1024, kH = 16, kDH = 64;

static __device__ __forceinline__ v8f wmma_bf16(v16bf a, v16bf b, v8f c) {
  // D = A(16x32 bf16) * B(32x16 bf16) + C(16x16 f32)
  return __builtin_amdgcn_wmma_f32_16x16x32_bf16(false, a, false, b, (short)0, c,
                                                 false, false);
}

// A-fragment (16x32 bf16): lane m = l&15; K contiguous [(l>>4)*16, +16)
static __device__ __forceinline__ v16bf ld_a_frag(const __bf16* p) {
  v8bf lo = *(const v8bf*)(p);
  v8bf hi = *(const v8bf*)(p + 8);
  v16bf r;
#pragma unroll
  for (int i = 0; i < 8; ++i) { r[i] = lo[i]; r[i + 8] = hi[i]; }
  return r;
}

// Tensor Data Mover: 2-D tile (bf16) global -> LDS, row-major compact in LDS.
// D# per CDNA5 ISA ch.8: group0 = {flags, lds_addr, global_addr, type=2},
// group1 = {wg_mask=0|data_size, tensor dims, tile dims, dim0 stride}.
// This toolchain exposes the 6-arg builtin: (g0, g1, g2, g3, g4, cpol).
static __device__ __forceinline__ void tdm_load_2d(unsigned lds_off,
                                                   const void* gptr,
                                                   unsigned tile0, unsigned tile1,
                                                   unsigned dim0, unsigned dim1,
                                                   unsigned stride0) {
  unsigned long long ga = (unsigned long long)(uintptr_t)gptr;
  v4u g0;
  g0[0] = 1u;                                         // count=1, user mode
  g0[1] = lds_off;                                    // lds_addr (bytes)
  g0[2] = (unsigned)ga;                               // global_addr[31:0]
  g0[3] = (unsigned)((ga >> 32) & 0x01ffffffull) | 0x80000000u; // [56:32]|type=2
  v8i_ g1;
  g1[0] = (int)(1u << 16);                            // data_size=1 -> 2 bytes
  g1[1] = (int)((dim0 & 0xffffu) << 16);              // tensor_dim0[15:0]
  g1[2] = (int)(((dim0 >> 16) & 0xffffu) | ((dim1 & 0xffffu) << 16));
  g1[3] = (int)(((dim1 >> 16) & 0xffffu) | ((tile0 & 0xffffu) << 16));
  g1[4] = (int)(tile1 & 0xffffu);                     // tile_dim1, tile_dim2=0
  g1[5] = (int)stride0;                               // tensor_dim0_stride lo
  g1[6] = 0;
  g1[7] = 0;
  v4i_ z4 = {};
  v8i_ z8 = {};
  __builtin_amdgcn_tensor_load_to_lds(g0, g1, z4, z4, z8, 0);
}

__global__ void cvt_f32_bf16(const float* __restrict__ in, __bf16* __restrict__ out,
                             int n) {
  int i = blockIdx.x * blockDim.x + threadIdx.x;
  int stride = gridDim.x * blockDim.x;
  for (; i < n; i += stride) out[i] = (__bf16)in[i];
}

// C[M,N] = A[M,K](bf16 rm) @ Bt[N,K](bf16 rm)^T + bias[N]
// Workgroup: 8 waves = 128(M) x 64(N). B tile TDM-staged in LDS, double buffered.
template <bool F32OUT>
__global__ void __launch_bounds__(256)
gemm_bt_wmma(const __bf16* __restrict__ A, const __bf16* __restrict__ Bt,
             const float* __restrict__ bias, void* __restrict__ out,
             int M, int N, int K) {
  __shared__ __attribute__((aligned(32))) __bf16 bsh[2][64][32]; // [n][k]
  const int lane = threadIdx.x & 31;
  const int wslot = threadIdx.x >> 5;
  const int nbn = N >> 6;
  const int mb = blockIdx.x / nbn;
  const int nb = blockIdx.x % nbn;
  const int m0 = (mb << 7) + (wslot << 4);
  const int n0 = nb << 6;
  const int ln = lane & 15, lh = lane >> 4;

  v8f acc[4] = {};
  const __bf16* ap = A + (size_t)(m0 + ln) * K + (lh << 4);

  if (wslot == 0)
    tdm_load_2d((unsigned)(uintptr_t)&bsh[0][0][0], Bt + (size_t)n0 * K,
                /*tile0*/32, /*tile1*/64, /*dim0*/(unsigned)K, /*dim1*/(unsigned)N,
                /*stride*/(unsigned)K);
  int p = 0;
  for (int k0 = 0; k0 < K; k0 += 32) {
    if (wslot == 0) __builtin_amdgcn_s_wait_tensorcnt(0);
    __syncthreads();
    if (wslot == 0 && k0 + 32 < K)
      tdm_load_2d((unsigned)(uintptr_t)&bsh[p ^ 1][0][0],
                  Bt + (size_t)n0 * K + (k0 + 32), 32, 64, (unsigned)K,
                  (unsigned)N, (unsigned)K);
    v16bf a = ld_a_frag(ap + k0);
    __builtin_prefetch((const void*)(ap + k0 + 128), 0, 1);
#pragma unroll
    for (int j = 0; j < 4; ++j) {
      // B-fragment: lane = column n, 16 contiguous K values from LDS
      v16bf b = *(const v16bf*)&bsh[p][(j << 4) + ln][lh << 4];
      acc[j] = wmma_bf16(a, b, acc[j]);
    }
    __syncthreads();
    p ^= 1;
  }

#pragma unroll
  for (int j = 0; j < 4; ++j) {
#pragma unroll
    for (int r = 0; r < 8; ++r) {
      int m = m0 + (lh << 3) + r;
      int n = n0 + (j << 4) + ln;
      float v = acc[j][r] + bias[n];
      if constexpr (F32OUT) ((float*)out)[(size_t)m * N + n] = v;
      else                  ((__bf16*)out)[(size_t)m * N + n] = (__bf16)v;
    }
  }
}

// Batched [S,D] -> [D,S] transpose (bf16), 32x32 LDS tiles.
__global__ void transpose_bf16(const __bf16* __restrict__ in,
                               __bf16* __restrict__ out, int Sdim, int Ddim) {
  __shared__ __bf16 tile[32][33];
  const int b = blockIdx.z;
  const int d0 = blockIdx.x << 5, s0 = blockIdx.y << 5;
  const __bf16* ip = in  + (size_t)b * Sdim * Ddim;
  __bf16*       op = out + (size_t)b * Sdim * Ddim;
#pragma unroll
  for (int r = 0; r < 4; ++r) {
    int s = s0 + threadIdx.y + (r << 3);
    tile[threadIdx.y + (r << 3)][threadIdx.x] =
        ip[(size_t)s * Ddim + d0 + threadIdx.x];
  }
  __syncthreads();
#pragma unroll
  for (int r = 0; r < 4; ++r) {
    int d = d0 + threadIdx.y + (r << 3);
    op[(size_t)d * Sdim + s0 + threadIdx.x] = tile[threadIdx.x][threadIdx.y + (r << 3)];
  }
}

// Flash attention. Block = 8 waves sharing one (b,h); wave w owns q-tile
// qbase = qg*128 + w*16. K tile [32 kpos x 64 d] and V^T tile [64 dh x 32 kpos]
// are TDM-staged into LDS (double buffered) and shared by all 8 waves.
// Scores computed TRANSPOSED (St = K_tile @ Q^T) so a softmax row = lane pair.
__global__ void __launch_bounds__(256)
flash_attn_wmma(const __bf16* __restrict__ Q, const __bf16* __restrict__ K,
                const __bf16* __restrict__ Vt, const unsigned char* __restrict__ mask,
                __bf16* __restrict__ Ctx) {
  __shared__ __attribute__((aligned(32))) __bf16 ksh[2][32][64]; // [kpos][d]
  __shared__ __attribute__((aligned(32))) __bf16 vsh[2][64][32]; // [dh][kpos]
  __shared__ __attribute__((aligned(32))) __bf16 pbuf[8][32][16];
  const int lane = threadIdx.x & 31;
  const int wslot = threadIdx.x >> 5;
  const int qg = blockIdx.x & 15;          // 16 q-groups of 128 rows
  const int h  = (blockIdx.x >> 4) & 15;
  const int b  = blockIdx.x >> 8;

  const int qbase = (qg << 7) + (wslot << 4);
  const int ln = lane & 15, lh = lane >> 4;

  // Q^T as B-fragments: lane column q = ln, K(d) = lh*16 + i (contiguous d)
  v16bf qb[2];
  {
    const __bf16* qp = Q + (size_t)(b * kS + qbase + ln) * kD + h * kDH + (lh << 4);
#pragma unroll
    for (int kk = 0; kk < 2; ++kk) qb[kk] = *(const v16bf*)(qp + kk * 32);
  }

  v8f o[4] = {};                           // O^T tiles: M=dh(4x16), N=q(16)
  float mrow = -1e30f, ssum = 0.0f;

  const __bf16* kgp = K  + (size_t)b * kS * kD + h * kDH;       // rows: kpos (stride kD)
  const __bf16* vgp = Vt + (size_t)(b * kH + h) * kDH * kS;     // rows: dh (stride kS)
  const unsigned char* mbp = mask + (size_t)b * kS * kS + (size_t)(qbase + ln) * kS;

  if (wslot == 0) {
    tdm_load_2d((unsigned)(uintptr_t)&ksh[0][0][0], kgp, 64, 32, kD, kS, kD);
    tdm_load_2d((unsigned)(uintptr_t)&vsh[0][0][0], vgp, 32, 64, kS, kDH, kS);
  }
  int p = 0;
  for (int kb = 0; kb < kS; kb += 32) {
    if (wslot == 0) __builtin_amdgcn_s_wait_tensorcnt(0);
    __syncthreads();
    if (wslot == 0 && kb + 32 < kS) {
      tdm_load_2d((unsigned)(uintptr_t)&ksh[p ^ 1][0][0],
                  kgp + (size_t)(kb + 32) * kD, 64, 32, kD, kS, kD);
      tdm_load_2d((unsigned)(uintptr_t)&vsh[p ^ 1][0][0],
                  vgp + (kb + 32), 32, 64, kS, kDH, kS);
    }

    // St[kpos, q] for 32 key positions = two 16x16 tiles, K-dim = DH = 64
    v8f st[2];
#pragma unroll
    for (int t = 0; t < 2; ++t) {
      const __bf16* kp = &ksh[p][t * 16 + ln][0];
      v8f c = {};
      c = wmma_bf16(ld_a_frag(kp + (lh << 4)), qb[0], c);
      c = wmma_bf16(ld_a_frag(kp + 32 + (lh << 4)), qb[1], c);
      st[t] = c;
    }
    // scale, masked -> 1e-9 (reference semantics), tile max
    float tmax = -1e30f;
#pragma unroll
    for (int t = 0; t < 2; ++t) {
      unsigned long long mb =
          *(const unsigned long long*)(mbp + kb + t * 16 + (lh << 3));
#pragma unroll
      for (int r = 0; r < 8; ++r) {
        float v = st[t][r] * 0.125f;         // 1/sqrt(64)
        if (((mb >> (r * 8)) & 0xffULL) == 0ULL) v = 1e-9f;
        st[t][r] = v;
        tmax = fmaxf(tmax, v);
      }
    }
    tmax = fmaxf(tmax, __shfl_xor(tmax, 16, 32));
    float mnew  = fmaxf(mrow, tmax);
    float alpha = __expf(mrow - mnew);
    float lsum = 0.0f;
#pragma unroll
    for (int t = 0; t < 2; ++t)
#pragma unroll
      for (int r = 0; r < 8; ++r) {
        float pv = __expf(st[t][r] - mnew);
        st[t][r] = pv;
        lsum += pv;
      }
    lsum += __shfl_xor(lsum, 16, 32);
    ssum = ssum * alpha + lsum;
#pragma unroll
    for (int j = 0; j < 4; ++j)
#pragma unroll
      for (int r = 0; r < 8; ++r) o[j][r] *= alpha;

    // Re-lay P^T (C layout) into B-fragment order through per-wave LDS.
    __bf16* pl = &pbuf[wslot][0][0];
#pragma unroll
    for (int t = 0; t < 2; ++t)
#pragma unroll
      for (int r = 0; r < 8; ++r) {
        int k = t * 16 + (lh << 3) + r;
        int dstlane = ((k >> 4) << 4) + ln;
        pl[dstlane * 16 + (k & 15)] = (__bf16)st[t][r];
      }
    __builtin_amdgcn_wave_barrier();
    v16bf pfrag = *(const v16bf*)(pl + lane * 16);
    __builtin_amdgcn_wave_barrier();

    // O^T += Vt_tile(16 dh x 32 k) @ P^T(32 k x 16 q)
#pragma unroll
    for (int j = 0; j < 4; ++j) {
      const __bf16* vp = &vsh[p][j * 16 + ln][0];
      o[j] = wmma_bf16(ld_a_frag(vp + (lh << 4)), pfrag, o[j]);
    }
    mrow = mnew;
    __syncthreads();
    p ^= 1;
  }

  const float rinv = 1.0f / ssum;
#pragma unroll
  for (int j = 0; j < 4; ++j)
#pragma unroll
    for (int r = 0; r < 8; ++r) {
      int dh = j * 16 + (lh << 3) + r;
      Ctx[(size_t)(b * kS + qbase + ln) * kD + h * kDH + dh] =
          (__bf16)(o[j][r] * rinv);
    }
}

extern "C" void kernel_launch(void* const* d_in, const int* in_sizes, int n_in,
                              void* d_out, int out_size, void* d_ws, size_t ws_size,
                              hipStream_t stream) {
  const float* query = (const float*)d_in[0];
  const float* key   = (const float*)d_in[1];
  const float* value = (const float*)d_in[2];
  const unsigned char* mask = (const unsigned char*)d_in[3];
  const float* Wq = (const float*)d_in[4];
  const float* bq = (const float*)d_in[5];
  const float* Wk = (const float*)d_in[6];
  const float* bk = (const float*)d_in[7];
  const float* Wv = (const float*)d_in[8];
  const float* bv = (const float*)d_in[9];
  const float* Wo = (const float*)d_in[10];
  const float* bo = (const float*)d_in[11];
  float* out = (float*)d_out;

  const int XE = kB * kS * kD;   // 4,194,304 elements
  const int WE = kD * kD;        // 1,048,576 elements
  __bf16* w  = (__bf16*)d_ws;
  __bf16* xq = w;
  __bf16* xk = xq + XE;
  __bf16* xv = xk + XE;
  __bf16* wqb = xv + XE;
  __bf16* wkb = wqb + WE;
  __bf16* wvb = wkb + WE;
  __bf16* wob = wvb + WE;
  __bf16* Qb = wob + WE;
  __bf16* Kb = Qb + XE;
  __bf16* Vb = Kb + XE;
  __bf16* Vt = Vb + XE;
  __bf16* Cx = Vt + XE;

  cvt_f32_bf16<<<1024, 256, 0, stream>>>(query, xq, XE);
  cvt_f32_bf16<<<1024, 256, 0, stream>>>(key,   xk, XE);
  cvt_f32_bf16<<<1024, 256, 0, stream>>>(value, xv, XE);
  cvt_f32_bf16<<<1024, 256, 0, stream>>>(Wq, wqb, WE);
  cvt_f32_bf16<<<1024, 256, 0, stream>>>(Wk, wkb, WE);
  cvt_f32_bf16<<<1024, 256, 0, stream>>>(Wv, wvb, WE);
  cvt_f32_bf16<<<1024, 256, 0, stream>>>(Wo, wob, WE);

  const int M = kB * kS;                               // 4096 tokens
  const int gemm_blocks = (M / 128) * (kD / 64);       // 512 blocks of 8 waves

  gemm_bt_wmma<false><<<gemm_blocks, 256, 0, stream>>>(xq, wqb, bq, (void*)Qb, M, kD, kD);
  gemm_bt_wmma<false><<<gemm_blocks, 256, 0, stream>>>(xk, wkb, bk, (void*)Kb, M, kD, kD);
  gemm_bt_wmma<false><<<gemm_blocks, 256, 0, stream>>>(xv, wvb, bv, (void*)Vb, M, kD, kD);

  transpose_bf16<<<dim3(kD / 32, kS / 32, kB), dim3(32, 8), 0, stream>>>(Vb, Vt, kS, kD);

  // B*H*(S/128) = 512 blocks of 8 waves (one (b,h) per block)
  flash_attn_wmma<<<512, 256, 0, stream>>>(Qb, Kb, Vt, mask, Cx);

  gemm_bt_wmma<true><<<gemm_blocks, 256, 0, stream>>>(Cx, wob, bo, (void*)out, M, kD, kD);
}
